// LSTMMachine_9844065042735
// MI455X (gfx1250) — matrix operand order repaired and checked by reading, not compiled, not run on hardware
//
#include <hip/hip_runtime.h>

typedef __attribute__((ext_vector_type(16))) _Float16 v16h;
typedef __attribute__((ext_vector_type(8)))  float    v8f;
typedef __attribute__((ext_vector_type(4)))  float    f32x4;

#define BATCH 8192
#define SEQT  64
#define IFEAT 125     // input feature size (K), padded to 128
#define WB    32      // batch elements per workgroup
#define ROWS  (WB * SEQT)      // 2048 GEMM rows per workgroup
#define NT    256              // threads per block (8 waves)
#define NWAVE (NT / 32)
#define NTILE (ROWS / 16)      // 128 tiles of 16 rows
#define TPW   (NTILE / NWAVE)  // 16 tiles per wave

// Unconditional: 8 consecutive f32 -> f16 (caller guarantees c+8 <= IFEAT).
__device__ __forceinline__ void load8h_full(const float* __restrict__ p, int c, _Float16* d) {
  f32x4 x = *(const f32x4*)(p + c);
  f32x4 y = *(const f32x4*)(p + c + 4);
#pragma unroll
  for (int e = 0; e < 4; ++e) { d[e] = (_Float16)x[e]; d[4 + e] = (_Float16)y[e]; }
}

// Guarded variant (cold path: B staging only).
__device__ __forceinline__ void load8h_guard(const float* __restrict__ p, int c, _Float16* d) {
  if (c + 8 <= IFEAT) {
    load8h_full(p, c, d);
  } else {
#pragma unroll
    for (int e = 0; e < 8; ++e) d[e] = (c + e < IFEAT) ? (_Float16)p[c + e] : (_Float16)0.0f;
  }
}

__device__ __forceinline__ float sigmoidf_(float x) {
  return 1.0f / (1.0f + __expf(-x));   // exp limits give correct 0/1 saturation
}
__device__ __forceinline__ float tanhf_(float x) {
  x = fminf(fmaxf(x, -10.0f), 10.0f);  // avoid inf/inf
  float e = __expf(2.0f * x);
  return (e - 1.0f) / (e + 1.0f);
}

__global__ __launch_bounds__(NT) void lstm_fused_wmma(
    const float* __restrict__ image, const float* __restrict__ Wih,
    const float* __restrict__ Whh,   const float* __restrict__ bih,
    const float* __restrict__ bhh,   const float* __restrict__ fcw,
    const float* __restrict__ fcb,   float* __restrict__ out) {
  // t-major gates buffer: idx = (t*WB + bl)*8 + gate. Per-recurrence-lane stride is
  // 8 banks -> only 4-way LDS conflicts in phase 2 (vs 16-way for b-major).
  __shared__ float gates[ROWS * 8];  // 64 KB

  const int tid  = threadIdx.x;
  const int wave = tid >> 5;
  const int lane = tid & 31;
  const int hi   = lane >> 4;   // half-wave select
  const int n    = lane & 15;   // N column (gate; valid 0..7) / M row for A loads
  const int b0   = blockIdx.x * WB;

  // ---- Stage B = Wih^T (K padded 125->128, N padded 8->16) once per wave.
  // B layout: lanes 0-15 hold K {k0+0..7, k0+16..23}; lanes 16-31 hold K {k0+8..15, k0+24..31}.
  v16h Bm[4];
  {
    const float* wrow = Wih + (size_t)((n < 8) ? n : 0) * IFEAT;
#pragma unroll
    for (int kk = 0; kk < 4; ++kk) {
      int c = kk * 32 + hi * 8;
      _Float16 t16[16];
      load8h_guard(wrow, c, t16);
      load8h_guard(wrow, c + 16, t16 + 8);
      v16h b;
#pragma unroll
      for (int e = 0; e < 16; ++e) b[e] = (n < 8) ? t16[e] : (_Float16)0.0f;
      Bm[kk] = b;
    }
  }

  // ---- Phase 1: gates_x = image_block * Wih^T via v_wmma_f32_16x16x32_f16.
  // A layout: lanes 0-15 (M=row) hold K {k0+0..7, k0+16..23}; lanes 16-31 hold
  // K {k0+8..15, k0+24..31}.
  for (int tile = wave * TPW; tile < (wave + 1) * TPW; ++tile) {
    int row = tile * 16 + n;  // local row = bl*64 + t (b-major => contiguous image reads)
    const float* rowp =
        image + ((size_t)(b0 + (row >> 6)) * SEQT + (size_t)(row & 63)) * IFEAT;
    v8f acc = {};
#pragma unroll
    for (int kk = 0; kk < 4; ++kk) {
      int c = kk * 32 + hi * 8;
      _Float16 t16[16];
      load8h_full(rowp, c, t16);              // max c=104 -> ends at 112 <= 125: always safe
      if (kk < 3 || !hi) {
        load8h_full(rowp, c + 16, t16 + 8);   // max start 112 -> ends at 120 <= 125: safe
      } else {
        // kk==3, hi: K = 120..127 -> floats 120..124 valid, pad 125..127 with 0
        f32x4 x = *(const f32x4*)(rowp + 120);
        t16[8]  = (_Float16)x[0]; t16[9]  = (_Float16)x[1];
        t16[10] = (_Float16)x[2]; t16[11] = (_Float16)x[3];
        t16[12] = (_Float16)rowp[124];
        t16[13] = (_Float16)0.0f; t16[14] = (_Float16)0.0f; t16[15] = (_Float16)0.0f;
      }
      v16h a;
#pragma unroll
      for (int e = 0; e < 16; ++e) a[e] = t16[e];
      acc = __builtin_amdgcn_wmma_f32_16x16x32_f16(false, a, false, Bm[kk],
                                                   (short)0, acc, false, false);
    }
    // C layout: lane n = column, vgpr r -> M = r + 8*hi. Keep the 8 real gate columns.
    if (n < 8) {
      int m0 = tile * 16 + hi * 8;
#pragma unroll
      for (int r = 0; r < 8; ++r) {
        int rowc = m0 + r;                 // = bl*64 + t
        int bl = rowc >> 6, t = rowc & 63;
        gates[(t * WB + bl) * 8 + n] = acc[r];   // lanes differ in n -> conflict-free store
      }
    }
  }

  __syncthreads();

  // ---- Phase 2: sequential LSTM recurrence, one thread per batch element.
  if (tid < WB) {
    const int bl = tid;
    float w0[8], w1[8], bias[8];
#pragma unroll
    for (int k = 0; k < 8; ++k) {
      w0[k]   = Whh[k * 2 + 0];
      w1[k]   = Whh[k * 2 + 1];
      bias[k] = bih[k] + bhh[k];
    }
    float h0 = 0.f, h1 = 0.f, c0 = 0.f, c1 = 0.f;
    for (int t = 0; t < SEQT; ++t) {
      const float* gp = &gates[(t * WB + bl) * 8];
      float g[8];
#pragma unroll
      for (int k = 0; k < 8; ++k)
        g[k] = gp[k] + bias[k] + h0 * w0[k] + h1 * w1[k];
      // gate order: i, f, g, o (each of width H=2)
      float i0 = sigmoidf_(g[0]), i1 = sigmoidf_(g[1]);
      float f0 = sigmoidf_(g[2]), f1 = sigmoidf_(g[3]);
      float z0 = tanhf_(g[4]),    z1 = tanhf_(g[5]);
      float o0 = sigmoidf_(g[6]), o1 = sigmoidf_(g[7]);
      c0 = f0 * c0 + i0 * z0;
      c1 = f1 * c1 + i1 * z1;
      h0 = o0 * tanhf_(c0);
      h1 = o1 * tanhf_(c1);
    }
    out[b0 + bl] = h0 * fcw[0] + h1 * fcw[1] + fcb[0];
  }
}

extern "C" void kernel_launch(void* const* d_in, const int* in_sizes, int n_in,
                              void* d_out, int out_size, void* d_ws, size_t ws_size,
                              hipStream_t stream) {
  (void)in_sizes; (void)n_in; (void)d_ws; (void)ws_size; (void)out_size;
  const float* image = (const float*)d_in[0];
  const float* Wih   = (const float*)d_in[1];
  const float* Whh   = (const float*)d_in[2];
  const float* bih   = (const float*)d_in[3];
  const float* bhh   = (const float*)d_in[4];
  const float* fcw   = (const float*)d_in[5];
  const float* fcb   = (const float*)d_in[6];
  float* out = (float*)d_out;

  dim3 grid(BATCH / WB);   // 256 workgroups
  dim3 block(NT);          // 256 threads = 8 wave32
  hipLaunchKernelGGL(lstm_fused_wmma, grid, block, 0, stream,
                     image, Wih, Whh, bih, bhh, fcw, fcb, out);
}
